// GATLocationModel_47124381171767
// MI455X (gfx1250) — matrix operand order repaired and checked by reading, not compiled, not run on hardware
//
#include <hip/hip_runtime.h>
#include <hip/hip_bf16.h>

// GAT location model for MI455X (gfx1250), wave32.
// Dense projections: V_WMMA_F32_16X16X32_BF16 with hi/lo bf16 split (fp32-like accuracy).
// Graph softmax/aggregation: ordered-uint atomicMax + f32 atomicAdd scatter (bandwidth bound).

#define FIN   128
#define HEADS 4
#define HIDC  64

typedef __attribute__((ext_vector_type(16))) __bf16 v16bf;
typedef __attribute__((ext_vector_type(8)))  __bf16 v8bf;
typedef __attribute__((ext_vector_type(8)))  float  v8f;

static __device__ __forceinline__ unsigned short f32_to_bf16_rne(float f) {
    unsigned u = __float_as_uint(f);
    unsigned r = u + 0x7FFFu + ((u >> 16) & 1u);
    return (unsigned short)(r >> 16);
}
static __device__ __forceinline__ float bf16_to_f32(unsigned short h) {
    return __uint_as_float(((unsigned)h) << 16);
}
// order-preserving float<->uint for atomicMax
static __device__ __forceinline__ unsigned enc_ord(float f) {
    unsigned u = __float_as_uint(f);
    return (u & 0x80000000u) ? ~u : (u | 0x80000000u);
}
static __device__ __forceinline__ float dec_ord(unsigned u) {
    unsigned b = (u & 0x80000000u) ? (u ^ 0x80000000u) : ~u;
    return __uint_as_float(b);
}
static __device__ __forceinline__ float leaky02(float x) { return x > 0.f ? x : 0.2f * x; }

static __device__ __forceinline__ v16bf cat16(v8bf a, v8bf b) {
    return __builtin_shufflevector(a, b, 0,1,2,3,4,5,6,7,8,9,10,11,12,13,14,15);
}

// ---------------------------------------------------------------------------
// fp32 -> (bf16 hi, bf16 lo) split
__global__ void __launch_bounds__(256)
convert_split(const float* __restrict__ in, unsigned short* __restrict__ hi,
              unsigned short* __restrict__ lo, int n) {
    int i = blockIdx.x * blockDim.x + threadIdx.x;
    if (i >= n) return;
    float v = in[i];
    unsigned short h = f32_to_bf16_rne(v);
    hi[i] = h;
    lo[i] = f32_to_bf16_rne(v - bf16_to_f32(h));
}

// ---------------------------------------------------------------------------
// H[M,O] = X[M,F] @ W[O,F]^T ; one 16x16 tile per wave, hi/lo split -> 3 WMMAs/k-step.
__global__ void __launch_bounds__(32)
gemm_wmma_split(const unsigned short* __restrict__ ahi, const unsigned short* __restrict__ alo,
                const unsigned short* __restrict__ bhi, const unsigned short* __restrict__ blo,
                float* __restrict__ out, int F, int O) {
    const int lane = threadIdx.x;
    const int m0 = blockIdx.x << 4;
    const int o0 = blockIdx.y << 4;
    const int mrow = m0 + (lane & 15);
    const int ncol = o0 + (lane & 15);
    const int akb = (lane & 16) ? 16 : 0;   // A K-base (contiguous k = 16h + s, matches B)
    const int bkb = (lane & 16) ? 16 : 0;   // B K-base (contiguous k = 16h + s, matches A)
    const unsigned short* arh = ahi + (size_t)mrow * F;
    const unsigned short* arl = alo + (size_t)mrow * F;
    const unsigned short* brh = bhi + (size_t)ncol * F;   // B[k][n] = W[n*F+k]
    const unsigned short* brl = blo + (size_t)ncol * F;
    v8f acc = {};
    for (int k0 = 0; k0 < F; k0 += 32) {
        v8bf a0h = *(const v8bf*)(arh + k0 + akb);
        v8bf a1h = *(const v8bf*)(arh + k0 + akb + 8);
        v8bf a0l = *(const v8bf*)(arl + k0 + akb);
        v8bf a1l = *(const v8bf*)(arl + k0 + akb + 8);
        v8bf b0h = *(const v8bf*)(brh + k0 + bkb);
        v8bf b1h = *(const v8bf*)(brh + k0 + bkb + 8);
        v8bf b0l = *(const v8bf*)(brl + k0 + bkb);
        v8bf b1l = *(const v8bf*)(brl + k0 + bkb + 8);
        v16bf Ah = cat16(a0h, a1h), Al = cat16(a0l, a1l);
        v16bf Bh = cat16(b0h, b1h), Bl = cat16(b0l, b1l);
        acc = __builtin_amdgcn_wmma_f32_16x16x32_bf16(false, Al, false, Bh, (short)0, acc, false, false);
        acc = __builtin_amdgcn_wmma_f32_16x16x32_bf16(false, Ah, false, Bl, (short)0, acc, false, false);
        acc = __builtin_amdgcn_wmma_f32_16x16x32_bf16(false, Ah, false, Bh, (short)0, acc, false, false);
    }
    const int mbase = m0 + ((lane & 16) ? 8 : 0);
#pragma unroll
    for (int v = 0; v < 8; ++v)
        out[(size_t)(mbase + v) * O + ncol] = acc[v];
}

// ---------------------------------------------------------------------------
// alpha_s/alpha_d per (node, head); seed segment-max with the self-loop logit.
__global__ void __launch_bounds__(256)
alpha_init(const float* __restrict__ H, const float* __restrict__ a_src,
           const float* __restrict__ a_dst, int n_nodes, int heads,
           float* __restrict__ as, float* __restrict__ ad, unsigned* __restrict__ menc) {
    int idx = blockIdx.x * blockDim.x + threadIdx.x;
    if (idx >= n_nodes * heads) return;
    int n = idx / heads, h = idx - n * heads;
    const float* hp  = H + (size_t)n * heads * HIDC + h * HIDC;
    const float* asw = a_src + h * HIDC;
    const float* adw = a_dst + h * HIDC;
    float s = 0.f, d = 0.f;
#pragma unroll 8
    for (int c = 0; c < HIDC; ++c) { float v = hp[c]; s += v * asw[c]; d += v * adw[c]; }
    as[idx] = s; ad[idx] = d;
    menc[idx] = enc_ord(leaky02(s + d));  // self-loop seeds the max
}

// segment max over edges (atomicMax on ordered-uint encoding)
__global__ void __launch_bounds__(256)
edge_max(const int* __restrict__ src, const int* __restrict__ dst,
         const float* __restrict__ as, const float* __restrict__ ad,
         unsigned* __restrict__ menc, int nE, int heads) {
    int idx = blockIdx.x * blockDim.x + threadIdx.x;
    if (idx >= nE * heads) return;
    int e = idx / heads, h = idx - e * heads;
    int s = src[e], d = dst[e];
    float l = leaky02(as[s * heads + h] + ad[d * heads + h]);
    atomicMax(&menc[d * heads + h], enc_ord(l));
}

// seed sums + unnormalized accumulator with self-loop term (block = heads*64)
__global__ void
node_init(const float* __restrict__ H, const float* __restrict__ as,
          const float* __restrict__ ad, const unsigned* __restrict__ menc,
          float* __restrict__ sv, float* __restrict__ acc, int heads) {
    int n = blockIdx.x, t = threadIdx.x;
    int h = t >> 6, c = t & 63;
    float m = dec_ord(menc[n * heads + h]);
    float l = leaky02(as[n * heads + h] + ad[n * heads + h]);
    float p = __expf(l - m);
    if (c == 0) sv[n * heads + h] = p;
    size_t o = (size_t)n * heads * HIDC + t;
    acc[o] = H[o] * p;
}

// one wave per edge: scatter exp-weighted source features to destination
__global__ void __launch_bounds__(256)
edge_acc(const int* __restrict__ src, const int* __restrict__ dst,
         const float* __restrict__ as, const float* __restrict__ ad,
         const unsigned* __restrict__ menc, float* __restrict__ sv,
         const float* __restrict__ H, float* __restrict__ acc, int nE, int heads) {
    int wid  = (blockIdx.x * blockDim.x + threadIdx.x) >> 5;
    int lane = threadIdx.x & 31;
    if (wid >= nE) return;
    int s = src[wid], d = dst[wid];
    for (int h = 0; h < heads; ++h) {
        float l = leaky02(as[s * heads + h] + ad[d * heads + h]);
        float w = __expf(l - dec_ord(menc[d * heads + h]));
        if (lane == 0) atomicAdd(&sv[d * heads + h], w);
        const float* hp = H   + (size_t)s * heads * HIDC + h * HIDC;
        float*       ap = acc + (size_t)d * heads * HIDC + h * HIDC;
        atomicAdd(&ap[lane],      hp[lane]      * w);
        atomicAdd(&ap[lane + 32], hp[lane + 32] * w);
    }
}

// layers 1/2: normalize, +bias, ELU, emit next layer's bf16 hi/lo input (block = 256)
__global__ void __launch_bounds__(256)
finalize_elu_split(const float* __restrict__ acc, const float* __restrict__ sv,
                   const float* __restrict__ bias,
                   unsigned short* __restrict__ xhi, unsigned short* __restrict__ xlo) {
    int n = blockIdx.x, t = threadIdx.x;
    int h = t >> 6;
    float v = acc[(size_t)n * 256 + t] / sv[n * HEADS + h] + bias[t];
    v = v > 0.f ? v : (__expf(v) - 1.0f);
    unsigned short hi = f32_to_bf16_rne(v);
    size_t o = (size_t)n * 256 + t;
    xhi[o] = hi;
    xlo[o] = f32_to_bf16_rne(v - bf16_to_f32(hi));
}

// layer-3 finalize fused with output heads (one wave per node)
__global__ void __launch_bounds__(256)
head_out(const float* __restrict__ acc, const float* __restrict__ sv,
         const float* __restrict__ b3, const float* __restrict__ Waz,
         const float* __restrict__ baz, const float* __restrict__ Wd,
         const float* __restrict__ bd, float* __restrict__ out, int n_nodes) {
    int wid  = (blockIdx.x * blockDim.x + threadIdx.x) >> 5;
    int lane = threadIdx.x & 31;
    if (wid >= n_nodes) return;
    float inv = 1.0f / sv[wid];
    const float* ap = acc + (size_t)wid * HIDC;
    float v0 = ap[lane]      * inv + b3[lane];
    float v1 = ap[lane + 32] * inv + b3[lane + 32];
    float d0 = v0 * Waz[lane]      + v1 * Waz[lane + 32];
    float d1 = v0 * Waz[64 + lane] + v1 * Waz[96 + lane];
    float d2 = v0 * Wd[lane]       + v1 * Wd[lane + 32];
#pragma unroll
    for (int off = 16; off > 0; off >>= 1) {
        d0 += __shfl_xor(d0, off, 32);
        d1 += __shfl_xor(d1, off, 32);
        d2 += __shfl_xor(d2, off, 32);
    }
    if (lane == 0) {
        out[(size_t)wid * 2]     = tanhf(d0 + baz[0]);
        out[(size_t)wid * 2 + 1] = tanhf(d1 + baz[1]);
        out[(size_t)2 * n_nodes + wid] = 1.0f / (1.0f + __expf(-(d2 + bd[0])));
    }
}

// ---------------------------------------------------------------------------
extern "C" void kernel_launch(void* const* d_in, const int* in_sizes, int n_in,
                              void* d_out, int out_size, void* d_ws, size_t ws_size,
                              hipStream_t stream) {
    const float* x   = (const float*)d_in[0];
    const int*   ei  = (const int*)d_in[1];
    const float* W1  = (const float*)d_in[2];
    const float* a1s = (const float*)d_in[3];
    const float* a1d = (const float*)d_in[4];
    const float* b1  = (const float*)d_in[5];
    const float* W2  = (const float*)d_in[6];
    const float* a2s = (const float*)d_in[7];
    const float* a2d = (const float*)d_in[8];
    const float* b2  = (const float*)d_in[9];
    const float* W3  = (const float*)d_in[10];
    const float* a3s = (const float*)d_in[11];
    const float* a3d = (const float*)d_in[12];
    const float* b3  = (const float*)d_in[13];
    const float* Waz = (const float*)d_in[14];
    const float* baz = (const float*)d_in[15];
    const float* Wd  = (const float*)d_in[16];
    const float* bd  = (const float*)d_in[17];
    (void)n_in; (void)out_size; (void)ws_size;

    const int N = in_sizes[0] / FIN;   // 50000 (divisible by 16)
    const int E = in_sizes[1] / 2;     // 800000
    const int* esrc = ei;
    const int* edst = ei + E;

    // workspace carve (~157 MB), all slices 256B aligned
    char* p = (char*)d_ws;
    auto take = [&](size_t bytes) -> char* {
        char* r = p; p += (bytes + 255) & ~(size_t)255; return r;
    };
    const size_t NO = (size_t)N * 256;
    unsigned short* xhi = (unsigned short*)take(NO * 2);
    unsigned short* xlo = (unsigned short*)take(NO * 2);
    unsigned short* whi = (unsigned short*)take(256 * 256 * 2);
    unsigned short* wlo = (unsigned short*)take(256 * 256 * 2);
    float*    Hbuf = (float*)take(NO * 4);
    float*    Acc  = (float*)take(NO * 4);
    float*    As   = (float*)take((size_t)N * HEADS * 4);
    float*    Ad   = (float*)take((size_t)N * HEADS * 4);
    unsigned* Menc = (unsigned*)take((size_t)N * HEADS * 4);
    float*    Sv   = (float*)take((size_t)N * HEADS * 4);
    float*    outp = (float*)d_out;

    auto run_gat = [&](int F, const float* W, const float* asw, const float* adw, int heads) {
        const int O = heads * HIDC;
        convert_split<<<(F * O + 255) / 256, 256, 0, stream>>>(W, whi, wlo, F * O);
        dim3 g(N / 16, O / 16);
        gemm_wmma_split<<<g, 32, 0, stream>>>(xhi, xlo, whi, wlo, Hbuf, F, O);
        alpha_init<<<(N * heads + 255) / 256, 256, 0, stream>>>(Hbuf, asw, adw, N, heads, As, Ad, Menc);
        edge_max<<<(E * heads + 255) / 256, 256, 0, stream>>>(esrc, edst, As, Ad, Menc, E, heads);
        node_init<<<N, heads * HIDC, 0, stream>>>(Hbuf, As, Ad, Menc, Sv, Acc, heads);
        edge_acc<<<(E + 7) / 8, 256, 0, stream>>>(esrc, edst, As, Ad, Menc, Sv, Hbuf, Acc, E, heads);
    };

    // layer 1 (F=128, heads=4, concat)
    convert_split<<<(N * FIN + 255) / 256, 256, 0, stream>>>(x, xhi, xlo, N * FIN);
    run_gat(FIN, W1, a1s, a1d, HEADS);
    finalize_elu_split<<<N, 256, 0, stream>>>(Acc, Sv, b1, xhi, xlo);

    // layer 2 (F=256, heads=4, concat)
    run_gat(256, W2, a2s, a2d, HEADS);
    finalize_elu_split<<<N, 256, 0, stream>>>(Acc, Sv, b2, xhi, xlo);

    // layer 3 (F=256, heads=1, mean==identity) + output heads
    run_gat(256, W3, a3s, a3d, 1);
    head_out<<<(N + 7) / 8, 256, 0, stream>>>(Acc, Sv, b3, Waz, baz, Wd, bd, outp, N);
}
